// GAT_27848567947394
// MI455X (gfx1250) — compile-verified, matrix operand order checked by reading, not run on hardware
//
#include <hip/hip_runtime.h>

// ---------------------------------------------------------------------------
// Types for CDNA5 WMMA (wave32, 16x16x32 bf16 -> f32) and TDM descriptors
// ---------------------------------------------------------------------------
typedef __attribute__((ext_vector_type(8)))  float        v8f;
typedef __attribute__((ext_vector_type(4)))  __bf16       bf16x4;
typedef __attribute__((ext_vector_type(8)))  __bf16       bf16x8;
typedef __attribute__((ext_vector_type(16))) __bf16       bf16x16;
typedef __attribute__((ext_vector_type(4)))  unsigned int u32x4;
typedef __attribute__((ext_vector_type(4)))  int          i32x4;
typedef __attribute__((ext_vector_type(8)))  int          i32x8;

static __device__ __forceinline__ bf16x16 cat8(bf16x8 lo, bf16x8 hi) {
  bf16x16 r;
#pragma unroll
  for (int i = 0; i < 8; ++i) { r[i] = lo[i]; r[i + 8] = hi[i]; }
  return r;
}

// ---------------------------------------------------------------------------
// Tensor Data Mover: async-DMA one 2D bf16 tile (tileX x tileY, row stride
// strideX elements) from global to LDS. D# layout per CDNA5 ISA ch.8.
// Rows with Y >= tensorY are zero-filled (handles the M tail).
// ---------------------------------------------------------------------------
static __device__ __forceinline__ void tdm_load_2d_bf16(
    unsigned ldsOff, const __bf16* gptr, unsigned tileX, unsigned tileY,
    unsigned tensorX, unsigned tensorY, unsigned strideX)
{
  unsigned long long ga = (unsigned long long)gptr;
  u32x4 g0;
  g0[0] = 1u;                                               // count=1 (valid)
  g0[1] = ldsOff;                                           // lds_addr (bytes)
  g0[2] = (unsigned)(ga & 0xFFFFFFFFu);                     // global_addr lo
  g0[3] = (unsigned)((ga >> 32) & 0x01FFFFFFu) | (2u << 30);// addr hi | type=2
  i32x8 g1;
  g1[0] = (int)(1u << 16);                                  // data_size=2B
  g1[1] = (int)((tensorX & 0xFFFFu) << 16);                 // tensor_dim0 lo16
  g1[2] = (int)(((tensorX >> 16) & 0xFFFFu) | ((tensorY & 0xFFFFu) << 16));
  g1[3] = (int)(((tensorY >> 16) & 0xFFFFu) | ((tileX & 0xFFFFu) << 16));
  g1[4] = (int)(tileY & 0xFFFFu);                           // tile_dim1, dim2=0
  g1[5] = (int)strideX;                                     // dim0 stride lo32
  g1[6] = 0;                                                // stride hi / dim1 stride
  g1[7] = 0;
  i32x4 z4 = {0, 0, 0, 0};
  i32x8 z8 = {0, 0, 0, 0, 0, 0, 0, 0};
  __builtin_amdgcn_tensor_load_to_lds(g0, g1, z4, z4, z8, 0);
}

// ---------------------------------------------------------------------------
// GEMM: C[M,N] = A[M,K] * W[K,N], K == 256, A bf16 row-major [M,K],
// W pre-transposed bf16 [N,K]. f32 accumulate via WMMA.
// Block: 256 threads = 8 waves; block tile 128x64, K-step 32.
// Tiles arrive in LDS via TDM (wave 0 issues, s_wait_tensorcnt, barrier).
// ---------------------------------------------------------------------------
#define GEMM_K 256
#define TM 128
#define TN 64
#define TK 32

__global__ __launch_bounds__(256)
void gat_gemm_bf16(const __bf16* __restrict__ A, const __bf16* __restrict__ Wt,
                   float* __restrict__ C, int M, int N)
{
  __shared__ __bf16 As[TM][TK];   // 8 KB, As[r][k] = A[rowBlk+r][k0+k]
  __shared__ __bf16 Bt[TN][TK];   // 4 KB, Bt[n][k] = W[k0+k][colBlk+n]

  const int tid  = threadIdx.x;
  const int lane = tid & 31;
  const int wave = tid >> 5;
  const int rowBlk = blockIdx.x * TM;
  const int colBlk = blockIdx.y * TN;

  const unsigned ldsA = (unsigned)(unsigned long long)(uintptr_t)(&As[0][0]);
  const unsigned ldsB = (unsigned)(unsigned long long)(uintptr_t)(&Bt[0][0]);

  v8f acc[4];
#pragma unroll
  for (int t = 0; t < 4; ++t) acc[t] = (v8f){0.f,0.f,0.f,0.f,0.f,0.f,0.f,0.f};

  const int mr = lane & 15;          // row in wave tile / col of B tile
  const int ks = (lane >> 4) << 3;   // 0 or 8: K sub-offset per ISA A/B layout

  for (int k0 = 0; k0 < GEMM_K; k0 += TK) {
    if (tid < 32) {                  // wave 0 drives the Tensor Data Mover
      tdm_load_2d_bf16(ldsA, A + (size_t)rowBlk * GEMM_K + k0,
                       TK, TM, TK, (unsigned)(M - rowBlk), GEMM_K);
      tdm_load_2d_bf16(ldsB, Wt + (size_t)colBlk * GEMM_K + k0,
                       TK, TN, TK, (unsigned)(N - colBlk), GEMM_K);
      __builtin_amdgcn_s_wait_tensorcnt(0);
    }
    __syncthreads();

    // A fragment: two ds_read_b128 per the 16-bit 16x32 A layout
    bf16x8 a_lo = *(const bf16x8*)(&As[wave * 16 + mr][ks]);
    bf16x8 a_hi = *(const bf16x8*)(&As[wave * 16 + mr][16 + ks]);
    bf16x16 afrag = cat8(a_lo, a_hi);

#pragma unroll
    for (int t = 0; t < 4; ++t) {
      bf16x8 b_lo = *(const bf16x8*)(&Bt[t * 16 + mr][ks]);
      bf16x8 b_hi = *(const bf16x8*)(&Bt[t * 16 + mr][16 + ks]);
      bf16x16 bfrag = cat8(b_lo, b_hi);
      acc[t] = __builtin_amdgcn_wmma_f32_16x16x32_bf16(
          false, afrag, false, bfrag, (short)0, acc[t], false, false);
    }
    __syncthreads();
  }

  // C store per documented layout: VGPR r -> row r (lanes 0-15) / r+8 (16-31)
  const int cRow0 = rowBlk + wave * 16 + ((lane >> 4) << 3);
  const int cCol  = colBlk + (lane & 15);
#pragma unroll
  for (int t = 0; t < 4; ++t) {
#pragma unroll
    for (int r = 0; r < 8; ++r) {
      int gr = cRow0 + r;
      if (gr < M) C[(size_t)gr * N + cCol + t * 16] = acc[t][r];
    }
  }
}

// ---------------------------------------------------------------------------
// Prologue: f32 -> bf16 conversion (vectorized) and W transpose+convert
// ---------------------------------------------------------------------------
__global__ __launch_bounds__(256)
void conv_bf16(const float* __restrict__ in, __bf16* __restrict__ out,
               long long n4)
{
  long long i = (long long)blockIdx.x * blockDim.x + threadIdx.x;
  if (i >= n4) return;
  float4 v = ((const float4*)in)[i];
  bf16x4 o;
  o[0] = (__bf16)v.x; o[1] = (__bf16)v.y; o[2] = (__bf16)v.z; o[3] = (__bf16)v.w;
  ((bf16x4*)out)[i] = o;
}

__global__ __launch_bounds__(256)
void convT_w(const float* __restrict__ W, __bf16* __restrict__ Wt,
             int K, int nShift, long long total)
{
  long long i = (long long)blockIdx.x * blockDim.x + threadIdx.x;
  if (i >= total) return;
  int n = (int)(i & ((1 << nShift) - 1));
  int k = (int)(i >> nShift);
  Wt[(size_t)n * K + k] = (__bf16)W[((size_t)k << nShift) + n];
}

// ---------------------------------------------------------------------------
// Attention scores: one wave per (node, head); e_src/e_dst = <h[n,h,:], a>
// ---------------------------------------------------------------------------
__global__ __launch_bounds__(256)
void gat_scores(const float* __restrict__ h, const float* __restrict__ a_src,
                const float* __restrict__ a_dst, float* __restrict__ eS,
                float* __restrict__ eD, long long totalNH, int hShift)
{
  long long gw = ((long long)blockIdx.x * blockDim.x + threadIdx.x) >> 5;
  int lane = threadIdx.x & 31;
  if (gw >= totalNH) return;
  int hh = (int)(gw & ((1 << hShift) - 1));
  long long n = gw >> hShift;
  int HD = 64 << hShift;
  const float* hp = h + n * HD + hh * 64;
  float s1 = 0.f, s2 = 0.f;
#pragma unroll
  for (int d = 0; d < 64; d += 32) {
    float v = hp[d + lane];
    s1 += v * a_src[hh * 64 + d + lane];
    s2 += v * a_dst[hh * 64 + d + lane];
  }
#pragma unroll
  for (int off = 16; off > 0; off >>= 1) {
    s1 += __shfl_down(s1, off, 32);
    s2 += __shfl_down(s2, off, 32);
  }
  if (lane == 0) { eS[gw] = s1; eD[gw] = s2; }
}

// ---------------------------------------------------------------------------
// Fills + edge softmax passes
// ---------------------------------------------------------------------------
__global__ __launch_bounds__(256)
void fill_u32(unsigned* __restrict__ p, unsigned v, long long n)
{
  long long i = (long long)blockIdx.x * blockDim.x + threadIdx.x;
  if (i < n) p[i] = v;
}

static __device__ __forceinline__ unsigned fenc(float f) {
  unsigned u = __float_as_uint(f);
  return (u & 0x80000000u) ? ~u : (u | 0x80000000u);
}

__global__ __launch_bounds__(256)
void edge_max(const int* __restrict__ src, const int* __restrict__ dst,
              const float* __restrict__ eS, const float* __restrict__ eD,
              unsigned* __restrict__ mEnc, long long E, int hShift)
{
  long long i = (long long)blockIdx.x * blockDim.x + threadIdx.x;
  if (i >= (E << hShift)) return;
  int hh = (int)(i & ((1 << hShift) - 1));
  long long e = i >> hShift;
  int s = src[e], d = dst[e];
  float v = eS[((long long)s << hShift) + hh] + eD[((long long)d << hShift) + hh];
  v = v > 0.f ? v : 0.2f * v;                      // LeakyReLU(0.2)
  atomicMax(&mEnc[((long long)d << hShift) + hh], fenc(v));
}

__global__ __launch_bounds__(256)
void m_finalize(float* __restrict__ m, long long n)
{
  long long i = (long long)blockIdx.x * blockDim.x + threadIdx.x;
  if (i >= n) return;
  unsigned u = ((const unsigned*)m)[i];
  float v = (u & 0x80000000u) ? __uint_as_float(u & 0x7FFFFFFFu)
                              : __uint_as_float(~u);
  unsigned eb = __float_as_uint(v) & 0x7F800000u;
  if (eb == 0x7F800000u) v = 0.f;                  // empty segment -> 0
  m[i] = v;
}

__global__ __launch_bounds__(256)
void edge_expsum(const int* __restrict__ src, const int* __restrict__ dst,
                 const float* __restrict__ eS, const float* __restrict__ eD,
                 const float* __restrict__ m, float* __restrict__ exch,
                 float* __restrict__ denom, long long E, int hShift)
{
  long long i = (long long)blockIdx.x * blockDim.x + threadIdx.x;
  if (i >= (E << hShift)) return;
  int hh = (int)(i & ((1 << hShift) - 1));
  long long e = i >> hShift;
  int s = src[e], d = dst[e];
  long long di = ((long long)d << hShift) + hh;
  float v = eS[((long long)s << hShift) + hh] + eD[di];
  v = v > 0.f ? v : 0.2f * v;
  float ex = __expf(v - m[di]);
  exch[i] = ex;
  atomicAdd(&denom[di], ex);
}

// One thread per (edge, 4-float chunk of H*D feats): gather h[src], atomic
// scatter into agg[dst]. D == 64 in all layers, so head = chunk >> 4.
__global__ __launch_bounds__(256)
void edge_aggregate(const int* __restrict__ src, const int* __restrict__ dst,
                    const float* __restrict__ hfeat, const float* __restrict__ exch,
                    const float* __restrict__ denom, float* __restrict__ agg,
                    long long E, int hShift, int chunkShift)
{
  long long i = (long long)blockIdx.x * blockDim.x + threadIdx.x;
  if (i >= (E << chunkShift)) return;
  long long e = i >> chunkShift;
  int c = (int)(i & ((1 << chunkShift) - 1));
  int hh = c >> 4;
  int s = src[e], d = dst[e];
  int HD = 64 << hShift;
  float alpha = exch[(e << hShift) + hh] /
                (denom[((long long)d << hShift) + hh] + 1e-16f);
  float4 hv = *(const float4*)(hfeat + (size_t)s * HD + (c << 2));
  float* op = agg + (size_t)d * HD + (c << 2);
  atomicAdd(op + 0, alpha * hv.x);
  atomicAdd(op + 1, alpha * hv.y);
  atomicAdd(op + 2, alpha * hv.z);
  atomicAdd(op + 3, alpha * hv.w);
}

// bias + ELU, emitting bf16 for the next layer's GEMM input
__global__ __launch_bounds__(256)
void bias_elu_bf16(const float* __restrict__ a, const float* __restrict__ b,
                   __bf16* __restrict__ o, long long n, int cmask)
{
  long long i = (long long)blockIdx.x * blockDim.x + threadIdx.x;
  if (i >= n) return;
  float v = a[i] + b[i & cmask];
  v = v > 0.f ? v : (__expf(v) - 1.f);
  o[i] = (__bf16)v;
}

// final bias (no activation), f32 output
__global__ __launch_bounds__(256)
void bias_out(const float* __restrict__ a, const float* __restrict__ b,
              float* __restrict__ o, long long n, int cmask)
{
  long long i = (long long)blockIdx.x * blockDim.x + threadIdx.x;
  if (i >= n) return;
  o[i] = a[i] + b[i & cmask];
}

// ---------------------------------------------------------------------------
// Driver
// ---------------------------------------------------------------------------
extern "C" void kernel_launch(void* const* d_in, const int* in_sizes, int n_in,
                              void* d_out, int out_size, void* d_ws, size_t ws_size,
                              hipStream_t stream)
{
  (void)n_in; (void)out_size; (void)ws_size;
  const float* x   = (const float*)d_in[0];
  const int*   eix = (const int*)  d_in[1];
  const float* Wm[3]  = { (const float*)d_in[2], (const float*)d_in[6],  (const float*)d_in[10] };
  const float* Asv[3] = { (const float*)d_in[3], (const float*)d_in[7],  (const float*)d_in[11] };
  const float* Adv[3] = { (const float*)d_in[4], (const float*)d_in[8],  (const float*)d_in[12] };
  const float* Bv[3]  = { (const float*)d_in[5], (const float*)d_in[9],  (const float*)d_in[13] };

  const long long Nn = in_sizes[0] / 256;
  const long long E  = in_sizes[1] / 2;
  const int* src = eix;
  const int* dst = eix + E;

  // Workspace arena
  float*  ws    = (float*)d_ws;
  float*  bufH  = ws;                    // Nn*256 f32 (h)
  float*  bufXA = bufH  + Nn * 256;      // Nn*256 f32 (agg)
  float*  eS    = bufXA + Nn * 256;      // Nn*4
  float*  eD    = eS + Nn * 4;           // Nn*4
  float*  mB    = eD + Nn * 4;           // Nn*4
  float*  dB    = mB + Nn * 4;           // Nn*4
  float*  exch  = dB + Nn * 4;           // E*4
  __bf16* xbf   = (__bf16*)(exch + E * 4);          // Nn*256 bf16 (GEMM input)
  __bf16* wtb   = xbf + Nn * 256;                   // 3 * 256*256 bf16 (W^T)
  __bf16* Wt[3] = { wtb, wtb + 65536, wtb + 2 * 65536 };

  // Prologue: bf16 activation copy + transposed bf16 weights
  long long n4 = Nn * 64;                // Nn*256 / 4
  conv_bf16<<<(unsigned)((n4 + 255) / 256), 256, 0, stream>>>(x, xbf, n4);
  const int nShifts[3] = {8, 8, 6};      // Nout = 256,256,64
  for (int L = 0; L < 3; ++L) {
    long long tot = 256LL << nShifts[L];
    convT_w<<<(unsigned)((tot + 255) / 256), 256, 0, stream>>>(
        Wm[L], Wt[L], 256, nShifts[L], tot);
  }

  const int Hs[3] = {4, 4, 1};
  for (int L = 0; L < 3; ++L) {
    const int H = Hs[L];
    const int hShift = (H == 4) ? 2 : 0;
    const int Nout = H * 64;
    const int chunkShift = (Nout == 256) ? 6 : 4;

    // 1) h = xbf @ W  (TDM tile loads + bf16 WMMA, f32 accumulate)
    dim3 gg((unsigned)((Nn + TM - 1) / TM), (unsigned)(Nout / TN), 1);
    gat_gemm_bf16<<<gg, 256, 0, stream>>>(xbf, Wt[L], bufH, (int)Nn, Nout);

    // 2) attention scores per (node, head)
    long long nh = Nn << hShift;
    gat_scores<<<(unsigned)((nh + 7) / 8), 256, 0, stream>>>(
        bufH, Asv[L], Adv[L], eS, eD, nh, hShift);

    // 3) softmax over incoming edges
    long long aggN = Nn * Nout;
    fill_u32<<<(unsigned)((nh + 255) / 256), 256, 0, stream>>>((unsigned*)mB, 0u, nh);
    fill_u32<<<(unsigned)((nh + 255) / 256), 256, 0, stream>>>((unsigned*)dB, 0u, nh);
    fill_u32<<<(unsigned)((aggN + 255) / 256), 256, 0, stream>>>((unsigned*)bufXA, 0u, aggN);

    long long eh = E << hShift;
    edge_max<<<(unsigned)((eh + 255) / 256), 256, 0, stream>>>(
        src, dst, eS, eD, (unsigned*)mB, E, hShift);
    m_finalize<<<(unsigned)((nh + 255) / 256), 256, 0, stream>>>(mB, nh);
    edge_expsum<<<(unsigned)((eh + 255) / 256), 256, 0, stream>>>(
        src, dst, eS, eD, mB, exch, dB, E, hShift);

    // 4) alpha-weighted aggregation
    long long ec = E << chunkShift;
    edge_aggregate<<<(unsigned)((ec + 255) / 256), 256, 0, stream>>>(
        src, dst, bufH, exch, dB, bufXA, E, hShift, chunkShift);

    // 5) bias (+ELU, bf16) for layers 0,1; final bias f32 for layer 2
    if (L < 2) {
      bias_elu_bf16<<<(unsigned)((aggN + 255) / 256), 256, 0, stream>>>(
          bufXA, Bv[L], xbf, aggN, Nout - 1);
    } else {
      bias_out<<<(unsigned)((aggN + 255) / 256), 256, 0, stream>>>(
          bufXA, Bv[L], (float*)d_out, aggN, Nout - 1);
    }
  }
}
